// MultiHeadAttention_80135499809057
// MI455X (gfx1250) — matrix operand order repaired and checked by reading, not compiled, not run on hardware
//
#include <hip/hip_runtime.h>

#define E_   1024
#define H_   16
#define DH_  64
#define B_   2
#define S_   2048
#define NTOK (B_ * S_)

typedef __attribute__((ext_vector_type(16))) __bf16 v16bf;
typedef __attribute__((ext_vector_type(8)))  __bf16 v8bf;
typedef __attribute__((ext_vector_type(8)))  float  v8f;

#define CAT8(l0, l1) __builtin_shufflevector(l0, l1, 0,1,2,3,4,5,6,7,8,9,10,11,12,13,14,15)

// ---------------------------------------------------------------------------
// Elementwise fp32 -> bf16 (8 per thread, vector store)
// ---------------------------------------------------------------------------
__global__ __launch_bounds__(256) void cvt_f32_bf16(const float* __restrict__ in,
                                                    __bf16* __restrict__ out, int n) {
  int i = (blockIdx.x * 256 + threadIdx.x) * 8;
  if (i < n) {
    v8bf o;
#pragma unroll
    for (int j = 0; j < 8; ++j) o[j] = (__bf16)in[i + j];
    *(v8bf*)&out[i] = o;
  }
}

// ---------------------------------------------------------------------------
// Wt[n][k] = (bf16) W[k][n]  (1024x1024), LDS-tiled transpose
// grid (32,32), block (32,8)
// ---------------------------------------------------------------------------
__global__ __launch_bounds__(256) void transpose_cvt(const float* __restrict__ W,
                                                     __bf16* __restrict__ Wt) {
  __shared__ __bf16 t[32][33];
  const int n0 = blockIdx.x * 32, k0 = blockIdx.y * 32;
  for (int j = threadIdx.y; j < 32; j += 8)
    t[j][threadIdx.x] = (__bf16)W[(k0 + j) * 1024 + n0 + threadIdx.x];
  __syncthreads();
  for (int j = threadIdx.y; j < 32; j += 8)
    Wt[(n0 + j) * 1024 + k0 + threadIdx.x] = t[threadIdx.x][j];
}

// ---------------------------------------------------------------------------
// C[M,1024] = A[M,1024](bf16) @ Wt[1024,1024]^T(bf16, K-major) + bias
// block = 256 threads = 8 waves as 2(M) x 4(N); wave tile 32x64
//   -> 8 x v_wmma_f32_16x16x32_bf16 per 32-wide K step, pure b128 feeds.
// MODE 0: fp32 [M][N]   (final output projection)
// MODE 1: bf16 [M][N]   (Q, K: fragment-friendly row-major per head)
// MODE 2: bf16 Vt[b*1024+n][s]  (V transposed per head for P@V B-fragments)
// grid: (1024/256, M/64)
// ---------------------------------------------------------------------------
template <int MODE>
__global__ __launch_bounds__(256) void gemm_bf16_wmma(
    const __bf16* __restrict__ A, const __bf16* __restrict__ Wt,
    const float* __restrict__ bias, void* __restrict__ Cout) {
  constexpr int K = 1024, N = 1024;
  const int tid = threadIdx.x, wave = tid >> 5, lane = tid & 31;
  const int hi = lane >> 4, lo = lane & 15;
  const int wm = wave & 1, wn = wave >> 1;
  const int rowBase = blockIdx.y * 64 + wm * 32;
  const int nBase   = blockIdx.x * 256 + wn * 64;

  v8f acc[2][4];
#pragma unroll
  for (int mt = 0; mt < 2; ++mt)
#pragma unroll
    for (int nt = 0; nt < 4; ++nt) {
      const float bval = bias[nBase + nt * 16 + lo];
#pragma unroll
      for (int r = 0; r < 8; ++r) acc[mt][nt][r] = bval;
    }

  for (int kb = 0; kb < K; kb += 32) {
    if (kb + 32 < K)
      __builtin_prefetch(&A[(rowBase + lo) * K + kb + 32], 0, 1);
    v16bf a[2], b[4];
#pragma unroll
    for (int mt = 0; mt < 2; ++mt) {
      const __bf16* ar = &A[(rowBase + mt * 16 + lo) * K + kb + hi * 16];
      v8bf l0 = *(const v8bf*)ar;
      v8bf l1 = *(const v8bf*)(ar + 8);
      a[mt] = CAT8(l0, l1);
    }
#pragma unroll
    for (int nt = 0; nt < 4; ++nt)
      b[nt] = *(const v16bf*)&Wt[(nBase + nt * 16 + lo) * K + kb + hi * 16];
#pragma unroll
    for (int mt = 0; mt < 2; ++mt)
#pragma unroll
      for (int nt = 0; nt < 4; ++nt)
        acc[mt][nt] = __builtin_amdgcn_wmma_f32_16x16x32_bf16(
            false, a[mt], false, b[nt], (short)0, acc[mt][nt], false, false);
  }

#pragma unroll
  for (int mt = 0; mt < 2; ++mt)
#pragma unroll
    for (int nt = 0; nt < 4; ++nt) {
      const int n = nBase + nt * 16 + lo;
      if constexpr (MODE == 0) {
        float* C = (float*)Cout;
#pragma unroll
        for (int r = 0; r < 8; ++r)
          C[(rowBase + mt * 16 + hi * 8 + r) * N + n] = acc[mt][nt][r];
      } else if constexpr (MODE == 1) {
        __bf16* C = (__bf16*)Cout;
#pragma unroll
        for (int r = 0; r < 8; ++r)
          C[(rowBase + mt * 16 + hi * 8 + r) * N + n] = (__bf16)acc[mt][nt][r];
      } else {  // MODE 2: Vt[(b*1024 + n)][s], 8 consecutive tokens -> one b128
        __bf16* C = (__bf16*)Cout;
        const int t0 = rowBase + mt * 16 + hi * 8;     // 8 consecutive tokens
        const int b = t0 >> 11, s0 = t0 & (S_ - 1);
        v8bf o;
#pragma unroll
        for (int r = 0; r < 8; ++r) o[r] = (__bf16)acc[mt][nt][r];
        *(v8bf*)&C[((size_t)(b * 1024 + n)) * S_ + s0] = o;
      }
    }
}

// ---------------------------------------------------------------------------
// Fused attention for one (b, h, 16-row q-tile).
// LDS: 128 KB fp32 logits + 64 KB bf16 probs + 2 KB async-staged Q tile.
// grid: (S/16, H, B), block 256 (8 waves)
// ---------------------------------------------------------------------------
__global__ __launch_bounds__(256) void attn_fused_kernel(
    const __bf16* __restrict__ Q, const __bf16* __restrict__ Km,
    const __bf16* __restrict__ Vt, const float* __restrict__ mask,
    float* __restrict__ attn, __bf16* __restrict__ ctx) {
  extern __shared__ char smem[];
  float*  p  = (float*)smem;                       // [16][S_] logits
  __bf16* pb = (__bf16*)(smem + 16 * S_ * 4);      // [16][S_] probs (bf16)
  __shared__ __bf16 qstage[16 * DH_];              // async-staged Q tile (2 KB)

  const int qt = blockIdx.x, h = blockIdx.y, b = blockIdx.z;
  const int qbase = qt * 16;
  const int tid = threadIdx.x, wave = tid >> 5, lane = tid & 31;
  const int hi = lane >> 4, lo = lane & 15;
  const long tokBase = (long)b * S_;

  // ---- async-stage the 16x64 Q tile into LDS (gfx1250 ASYNC path) ----
  // 128 lanes x 16 B; VDST = LDS byte offset, VADDR = 64-bit global address.
  if (wave < 4) {                                  // wave-uniform
    const int row = tid >> 3, chunk = tid & 7;     // 16 rows x 8 chunks
    const __bf16* gsrc = &Q[(tokBase + qbase + row) * E_ + h * DH_ + chunk * 8];
    unsigned ldst = (unsigned)(unsigned long long)(&qstage[row * DH_ + chunk * 8]);
    asm volatile("global_load_async_to_lds_b128 %0, %1, off"
                 :: "v"(ldst), "v"(gsrc) : "memory");
    asm volatile("s_wait_asynccnt 0" ::: "memory");
  }
  __syncthreads();

  // hoisted Q A-fragments (DH=64 = 2 K-chunks of 32) from LDS
  v16bf qa[2];
  {
    const __bf16* qrow = &qstage[lo * DH_];
#pragma unroll
    for (int c = 0; c < 2; ++c) {
      v8bf l0 = *(const v8bf*)(qrow + c * 32 + hi * 16);
      v8bf l1 = *(const v8bf*)(qrow + c * 32 + hi * 16 + 8);
      qa[c] = CAT8(l0, l1);
    }
  }

  // ---- phase 1: logits = QK^T * 1/8 + mask * -1e9 ----
  for (int kt = wave; kt < S_ / 16; kt += 8) {     // wave-uniform
    const int kv = kt * 16 + lo;
    const __bf16* krow = &Km[(tokBase + kv) * E_ + h * DH_];
    v8f acc = {};
#pragma unroll
    for (int c = 0; c < 2; ++c) {
      v16bf kf = *(const v16bf*)(krow + c * 32 + hi * 16);   // 32B contiguous
      acc = __builtin_amdgcn_wmma_f32_16x16x32_bf16(false, qa[c], false, kf,
                                                    (short)0, acc, false, false);
    }
    const float mbias = mask[b * S_ + kv] * -1e9f;
#pragma unroll
    for (int r = 0; r < 8; ++r)
      p[(hi * 8 + r) * S_ + kt * 16 + lo] = acc[r] * 0.125f + mbias;
  }
  __syncthreads();

  // ---- phase 2: softmax; each wave owns 2 rows; write attn + bf16 probs ----
#pragma unroll
  for (int rr = 0; rr < 2; ++rr) {
    const int row = wave * 2 + rr;
    float* pr = &p[row * S_];
    float mx = -3.4e38f;
    for (int c = lane; c < S_; c += 32) mx = fmaxf(mx, pr[c]);
#pragma unroll
    for (int off = 16; off >= 1; off >>= 1) mx = fmaxf(mx, __shfl_xor(mx, off, 32));
    float sum = 0.f;
    for (int c = lane; c < S_; c += 32) {
      float e = __expf(pr[c] - mx);
      pr[c] = e;
      sum += e;
    }
#pragma unroll
    for (int off = 16; off >= 1; off >>= 1) sum += __shfl_xor(sum, off, 32);
    const float inv = 1.0f / sum;
    float* arow = &attn[(((long)(b * H_ + h)) * S_ + (qbase + row)) * S_];
    __bf16* pbr = &pb[row * S_];
    for (int c = lane; c < S_; c += 32) {
      float v = pr[c] * inv;
      arow[c] = v;            // only global touch of attn: coalesced store
      pbr[c]  = (__bf16)v;    // bf16 copy for the P@V WMMA
    }
  }
  __syncthreads();

  // ---- phase 3: ctx = P @ V with ALL 8 waves ----
  // wave = (khalf 0..1) x (dslice 0..3): each wave 32 WMMAs (matches phase 1)
  {
    const int dbase = (wave & 3) * 16;
    const int khalf = wave >> 2;
    const __bf16* vrow = &Vt[((size_t)(b * 1024 + h * DH_ + dbase + lo)) * S_];
    v8f acc = {};
    const int k0 = khalf * (S_ / 2);
    for (int kb = k0; kb < k0 + S_ / 2; kb += 32) {
      v8bf l0 = *(const v8bf*)&pb[lo * S_ + kb + hi * 16];       // LDS b128
      v8bf l1 = *(const v8bf*)&pb[lo * S_ + kb + hi * 16 + 8];
      v16bf a = CAT8(l0, l1);
      v16bf bv = *(const v16bf*)(vrow + kb + hi * 16);           // 32B contiguous
      acc = __builtin_amdgcn_wmma_f32_16x16x32_bf16(false, a, false, bv,
                                                    (short)0, acc, false, false);
    }
    // stash 16x16 f32 partial in LDS (reuse dead fp32 logits region)
    float* part = p;                                // [8][16][16]
#pragma unroll
    for (int r = 0; r < 8; ++r)
      part[wave * 256 + (hi * 8 + r) * 16 + lo] = acc[r];
  }
  __syncthreads();

  // combine the two K-halves and store bf16 ctx (all 256 threads, 4 elems ea.)
  for (int i = tid; i < 16 * DH_; i += 256) {
    const int m = i >> 6, d = i & 63;
    const int w0 = d >> 4, c = (m & 7) * 16 + (d & 15) + (m >> 3) * 0;
    const float* part = p;
    float s = part[w0 * 256 + m * 16 + (d & 15)] +
              part[(w0 + 4) * 256 + m * 16 + (d & 15)];
    (void)c;
    ctx[(tokBase + qbase + m) * E_ + h * DH_ + d] = (__bf16)s;
  }
}

// ---------------------------------------------------------------------------
extern "C" void kernel_launch(void* const* d_in, const int* in_sizes, int n_in,
                              void* d_out, int out_size, void* d_ws, size_t ws_size,
                              hipStream_t stream) {
  const float* X    = (const float*)d_in[0];
  const float* mask = (const float*)d_in[1];
  const float* Wq   = (const float*)d_in[2];
  const float* bq   = (const float*)d_in[3];
  const float* Wk   = (const float*)d_in[4];
  const float* bk   = (const float*)d_in[5];
  const float* Wv   = (const float*)d_in[6];
  const float* bv   = (const float*)d_in[7];
  const float* Wo   = (const float*)d_in[8];
  const float* bo   = (const float*)d_in[9];

  float* out  = (float*)d_out;                      // [B,S,E]
  float* attn = out + (size_t)B_ * S_ * E_;         // [B,H,S,S]

  __bf16* Xb   = (__bf16*)d_ws;                     // 8 MB  [NTOK][E]
  __bf16* Wtq  = Xb  + (size_t)NTOK * E_;           // 2 MB each [N][K]
  __bf16* Wtk  = Wtq + (size_t)E_ * E_;
  __bf16* Wtv  = Wtk + (size_t)E_ * E_;
  __bf16* Wto  = Wtv + (size_t)E_ * E_;
  __bf16* Qb   = Wto + (size_t)E_ * E_;             // 8 MB  [NTOK][E]
  __bf16* Kb   = Qb  + (size_t)NTOK * E_;           // 8 MB
  __bf16* Vt   = Kb  + (size_t)NTOK * E_;           // 8 MB  [B*1024][S]
  __bf16* ctxb = Vt  + (size_t)NTOK * E_;           // 8 MB  [NTOK][E]

  // pre-convert activations and weights (amortized, L2-resident afterwards)
  cvt_f32_bf16<<<(NTOK * E_) / (256 * 8), 256, 0, stream>>>(X, Xb, NTOK * E_);
  dim3 tb(32, 8), tg(32, 32);
  transpose_cvt<<<tg, tb, 0, stream>>>(Wq, Wtq);
  transpose_cvt<<<tg, tb, 0, stream>>>(Wk, Wtk);
  transpose_cvt<<<tg, tb, 0, stream>>>(Wv, Wtv);
  transpose_cvt<<<tg, tb, 0, stream>>>(Wo, Wto);

  dim3 ggrid(1024 / 256, NTOK / 64);                // (4, 64)
  gemm_bf16_wmma<1><<<ggrid, 256, 0, stream>>>(Xb, Wtq, bq, (void*)Qb);
  gemm_bf16_wmma<1><<<ggrid, 256, 0, stream>>>(Xb, Wtk, bk, (void*)Kb);
  gemm_bf16_wmma<2><<<ggrid, 256, 0, stream>>>(Xb, Wtv, bv, (void*)Vt);

  attn_fused_kernel<<<dim3(S_ / 16, H_, B_), 256,
                      16 * S_ * 4 + 16 * S_ * 2, stream>>>(Qb, Kb, Vt, mask,
                                                           attn, ctxb);

  gemm_bf16_wmma<0><<<ggrid, 256, 0, stream>>>(ctxb, Wto, bo, (void*)out);
}
